// ParaformerWithHotword_23948737642801
// MI455X (gfx1250) — compile-verified, matrix operand order checked by reading, not run on hardware
//
#include <hip/hip_runtime.h>

// ---------------------------------------------------------------------------
// Types
// ---------------------------------------------------------------------------
typedef _Float16 h16;
typedef __attribute__((ext_vector_type(8)))  _Float16 v8h;
typedef __attribute__((ext_vector_type(16))) _Float16 v16h;
typedef __attribute__((ext_vector_type(8)))  float    v8f;
// GCC-style 4xint vector: exact pointee type of the async-LDS builtins
// (diagnostic showed '__attribute__((__vector_size__(4 * sizeof(int)))) int').
typedef int v4ig __attribute__((vector_size(16)));

#define DEVI __device__ __forceinline__

#define AS1 __attribute__((address_space(1)))
#define AS3 __attribute__((address_space(3)))

// gfx1250 async global->LDS copy path (ASYNCcnt), guarded for toolchain support.
#if defined(__has_builtin)
#if __has_builtin(__builtin_amdgcn_global_load_async_to_lds_b128) && \
    __has_builtin(__builtin_amdgcn_s_wait_asynccnt)
#define USE_ASYNC_LDS 1
#endif
#endif
#ifndef USE_ASYNC_LDS
#define USE_ASYNC_LDS 0
#endif

DEVI int imin(int a, int b) { return a < b ? a : b; }

// ---------------------------------------------------------------------------
// Wave32 / block reductions
// ---------------------------------------------------------------------------
DEVI float warp_sum(float v) {
  for (int o = 16; o; o >>= 1) v += __shfl_xor(v, o, 32);
  return v;
}
DEVI float warp_max(float v) {
  for (int o = 16; o; o >>= 1) v = fmaxf(v, __shfl_xor(v, o, 32));
  return v;
}

__device__ float block_sum(float v) {
  __shared__ float sh[33];
  int lane = threadIdx.x & 31, w = threadIdx.x >> 5;
  int nw = ((int)blockDim.x + 31) >> 5;
  v = warp_sum(v);
  __syncthreads();
  if (lane == 0) sh[w] = v;
  __syncthreads();
  if (w == 0) {
    float t = (lane < nw) ? sh[lane] : 0.f;
    t = warp_sum(t);
    if (lane == 0) sh[32] = t;
  }
  __syncthreads();
  float r = sh[32];
  __syncthreads();
  return r;
}

__device__ float block_max(float v) {
  __shared__ float sh[33];
  int lane = threadIdx.x & 31, w = threadIdx.x >> 5;
  int nw = ((int)blockDim.x + 31) >> 5;
  v = warp_max(v);
  __syncthreads();
  if (lane == 0) sh[w] = v;
  __syncthreads();
  if (w == 0) {
    float t = (lane < nw) ? sh[lane] : -1e30f;
    t = warp_max(t);
    if (lane == 0) sh[32] = t;
  }
  __syncthreads();
  float r = sh[32];
  __syncthreads();
  return r;
}

// Sinusoidal positional encoding element.
DEVI float pe_val(int pos, int c, int D) {
  float dv = expf((float)(c & ~1) * (-9.210340372f / (float)D)); // ln(10000)
  float arg = (float)pos * dv;
  return (c & 1) ? cosf(arg) : sinf(arg);
}

// Build a 16x32 f16 WMMA fragment row from LDS (ISA A/B VGPR layout):
// lanes 0-15 take K=0..7 and 16..23, lanes 16-31 take K=8..15 and 24..31.
DEVI v16h mk_frag(const h16* row, int hs) {
  v8h a0 = *(const v8h*)(row + hs * 8);
  v8h a1 = *(const v8h*)(row + 16 + hs * 8);
  return __builtin_shufflevector(a0, a1, 0, 1, 2, 3, 4, 5, 6, 7,
                                 8, 9, 10, 11, 12, 13, 14, 15);
}

// ---------------------------------------------------------------------------
// WMMA GEMM:  C[M,N] = act(A[M,K] * Bt[N,K]^T + bias)
//   A, Bt f16; C f32; optional f16 mirror Ch. Batched via blockIdx.z with
//   element strides sA/sB/sC/sCh. 128 threads = 4 waves. Block tile 128x64;
//   each wave computes 32(M) x 64(N): 2 A-fragments x 4 B-fragments ->
//   8 independent v_wmma_f32_16x16x32_f16 per 32-K step (no RAW hazards).
//   LDS tiles are double-buffered; staging uses the CDNA5 async
//   global->LDS copy engine when available (ASYNCcnt + s_wait_asynccnt).
//   Requires K % 32 == 0 (all f16 operands are built zero-padded).
// ---------------------------------------------------------------------------
__global__ __launch_bounds__(128) void k_gemm(
    const h16* __restrict__ A, int lda, long sA,
    const h16* __restrict__ B, int ldb, long sB,
    const float* __restrict__ bias,
    float* __restrict__ C, int ldc, long sC,
    h16* __restrict__ Ch, int ldch, long sCh,
    int M, int N, int K, int act) {
  __shared__ h16 As[2][128][32];
  __shared__ h16 Bs[2][64][32];

  const int tid = threadIdx.x;
  const int wave = tid >> 5;
  const int lane = tid & 31;
  const int hs = lane >> 4;   // K-half selector (ISA A/B layout)
  const int rw = lane & 15;   // row-within-16
  const int m0 = blockIdx.y * 128;
  const int n0 = blockIdx.x * 64;

  A += (long)blockIdx.z * sA;
  B += (long)blockIdx.z * sB;

  v8f acc[8] = {};

  auto stage = [&](int buf, int k0) {
    // A tile: 128x32 halves = 512 x 16B chunks; 4 per thread.
    for (int c = tid; c < 512; c += 128) {
      int r = c >> 2, off = (c & 3) * 8;
      int gr = imin(m0 + r, M - 1);
      const h16* src = &A[(long)gr * lda + k0 + off];
      h16* dst = &As[buf][r][off];
#if USE_ASYNC_LDS
      __builtin_amdgcn_global_load_async_to_lds_b128(
          (AS1 v4ig*)(v4ig*)(void*)src, (AS3 v4ig*)(v4ig*)(void*)dst, 0, 0);
#else
      *(v8h*)dst = *(const v8h*)src;
#endif
    }
    // B tile: 64x32 halves = 256 x 16B chunks; 2 per thread.
    for (int c = tid; c < 256; c += 128) {
      int r = c >> 2, off = (c & 3) * 8;
      int gr = imin(n0 + r, N - 1);
      const h16* src = &B[(long)gr * ldb + k0 + off];
      h16* dst = &Bs[buf][r][off];
#if USE_ASYNC_LDS
      __builtin_amdgcn_global_load_async_to_lds_b128(
          (AS1 v4ig*)(v4ig*)(void*)src, (AS3 v4ig*)(v4ig*)(void*)dst, 0, 0);
#else
      *(v8h*)dst = *(const v8h*)src;
#endif
    }
    if (k0 + 32 < K) {  // hint K+2 tile toward L2 (global_prefetch_b8)
      __builtin_prefetch(&A[(long)imin(m0 + (tid >> 2), M - 1) * lda + k0 + 32], 0, 1);
      __builtin_prefetch(&B[(long)imin(n0 + (tid >> 2), N - 1) * ldb + k0 + 32], 0, 1);
    }
  };

  stage(0, 0);
#if USE_ASYNC_LDS
  __builtin_amdgcn_s_wait_asynccnt(0);
#endif
  __syncthreads();

  const int nk = K >> 5;
  for (int ki = 0; ki < nk; ++ki) {
    const int cur = ki & 1;
    if (ki + 1 < nk) stage(cur ^ 1, (ki + 1) * 32);  // overlap with WMMAs below

    v16h af0 = mk_frag(&As[cur][wave * 32 + rw][0], hs);
    v16h af1 = mk_frag(&As[cur][wave * 32 + 16 + rw][0], hs);
#pragma unroll
    for (int j = 0; j < 4; ++j) {
      v16h bf = mk_frag(&Bs[cur][j * 16 + rw][0], hs);
      acc[j] = __builtin_amdgcn_wmma_f32_16x16x32_f16(
          false, af0, false, bf, (short)0, acc[j], false, false);
      acc[4 + j] = __builtin_amdgcn_wmma_f32_16x16x32_f16(
          false, af1, false, bf, (short)0, acc[4 + j], false, false);
    }
#if USE_ASYNC_LDS
    if (ki + 1 < nk) __builtin_amdgcn_s_wait_asynccnt(0);
#endif
    __syncthreads();
  }

  // D layout: lanes 0-15 -> N=lane, M=v; lanes 16-31 -> N=lane-16, M=v+8.
#pragma unroll
  for (int sub = 0; sub < 2; ++sub) {
#pragma unroll
    for (int j = 0; j < 4; ++j) {
#pragma unroll
      for (int v = 0; v < 8; ++v) {
        int gm = m0 + wave * 32 + sub * 16 + v + 8 * hs;
        int gn = n0 + j * 16 + rw;
        if (gm < M && gn < N) {
          float x = acc[sub * 4 + j][v];
          if (bias) x += bias[gn];
          if (act == 1) x = fmaxf(x, 0.f);
          if (C)  C[(long)blockIdx.z * sC + (long)gm * ldc + gn] = x;
          if (Ch) Ch[(long)blockIdx.z * sCh + (long)gm * ldch + gn] = (h16)x;
        }
      }
    }
  }
}

// ---------------------------------------------------------------------------
// Conversion / transpose helpers (produce zero-padded f16 operands)
// ---------------------------------------------------------------------------
__global__ void k_cvt(const float* __restrict__ s, h16* __restrict__ d, long n) {
  long i = (long)blockIdx.x * blockDim.x + threadIdx.x;
  if (i < n) d[i] = (h16)s[i];
}

// row-major [rows,K] f32 -> [rows,Kp] f16 zero-padded
__global__ void k_cvt_pad(const float* __restrict__ s, h16* __restrict__ d,
                          int K, int Kp, long n) {
  long i = (long)blockIdx.x * blockDim.x + threadIdx.x;
  if (i >= n) return;
  long r = i / Kp; int k = (int)(i % Kp);
  d[i] = (k < K) ? (h16)s[r * K + k] : (h16)0.f;
}

// W[K,N] f32 -> Wt[N,Kp] f16 (transpose + convert + zero pad)
__global__ void k_tcvt(const float* __restrict__ W, h16* __restrict__ Wt,
                       int K, int N, int Kp, long n) {
  long i = (long)blockIdx.x * blockDim.x + threadIdx.x;
  if (i >= n) return;
  long nn = i / Kp; int k = (int)(i % Kp);
  Wt[i] = (k < K) ? (h16)W[(long)k * N + nn] : (h16)0.f;
}

// x += PE(row % pemod); write f16 mirror
__global__ void k_add_pe_cvt(float* __restrict__ x, h16* __restrict__ xh,
                             int D, int pemod) {
  long row = blockIdx.x;
  int pos = (int)(row % pemod);
  for (int c = threadIdx.x; c < D; c += blockDim.x) {
    float v = x[row * D + c] + pe_val(pos, c, D);
    x[row * D + c] = v;
    xh[row * D + c] = (h16)v;
  }
}

// ---------------------------------------------------------------------------
// LayerNorm(residual + y) with f16 mirror
// ---------------------------------------------------------------------------
__global__ void k_ln_res(const float* __restrict__ res, const float* __restrict__ y,
                         const float* __restrict__ g, const float* __restrict__ b,
                         float* __restrict__ out, h16* __restrict__ outh, int D) {
  long row = blockIdx.x;
  const float* rr = res + row * D;
  const float* yy = y + row * D;
  float s = 0.f;
  for (int c = threadIdx.x; c < D; c += blockDim.x) s += rr[c] + yy[c];
  float mean = block_sum(s) / (float)D;
  float v = 0.f;
  for (int c = threadIdx.x; c < D; c += blockDim.x) {
    float d = rr[c] + yy[c] - mean;
    v += d * d;
  }
  float var = block_sum(v) / (float)D;
  float inv = rsqrtf(var + 1e-5f);
  for (int c = threadIdx.x; c < D; c += blockDim.x) {
    float val = (rr[c] + yy[c] - mean) * inv * g[c] + b[c];
    out[row * D + c] = val;
    if (outh) outh[row * D + c] = (h16)val;
  }
}

// ---------------------------------------------------------------------------
// Softmax over N (scaled), writing f16 probabilities zero-padded to ldP.
// grid = (rows, batch)
// ---------------------------------------------------------------------------
__global__ void k_softmax(const float* __restrict__ S, long sS, int ldS,
                          h16* __restrict__ P, long sP, int ldP,
                          int N, float scale) {
  const float* srow = S + (long)blockIdx.y * sS + (long)blockIdx.x * ldS;
  h16* prow = P + (long)blockIdx.y * sP + (long)blockIdx.x * ldP;
  float mx = -1e30f;
  for (int i = threadIdx.x; i < N; i += blockDim.x)
    mx = fmaxf(mx, srow[i] * scale);
  mx = block_max(mx);
  float sum = 0.f;
  for (int i = threadIdx.x; i < N; i += blockDim.x)
    sum += expf(srow[i] * scale - mx);
  sum = block_sum(sum);
  float inv = 1.f / (sum + 1e-30f);
  for (int i = threadIdx.x; i < ldP; i += blockDim.x)
    prow[i] = (i < N) ? (h16)(expf(srow[i] * scale - mx) * inv) : (h16)0.f;
}

// ---------------------------------------------------------------------------
// Predictor conv1d (kernel 3, pad 1) + ReLU.  in:[T,Cin]  w:[Co,Cin,3]  out:[T,Co]
// ---------------------------------------------------------------------------
__global__ void k_conv1d_relu(const float* __restrict__ in, const float* __restrict__ w,
                              const float* __restrict__ b, float* __restrict__ out,
                              int T, int Cin, int Co) {
  long idx = (long)blockIdx.x * blockDim.x + threadIdx.x;
  if (idx >= (long)T * Co) return;
  int t = (int)(idx / Co), co = (int)(idx % Co);
  float s = b[co];
  for (int j = 0; j < 3; ++j) {
    int tt = t + j - 1;
    if (tt < 0 || tt >= T) continue;
    const float* ip = in + (long)tt * Cin;
    const float* wp = w + (long)co * Cin * 3 + j;
    for (int ci = 0; ci < Cin; ++ci) s += ip[ci] * wp[(long)ci * 3];
  }
  out[(long)t * Co + co] = fmaxf(s, 0.f);
}

// out[row] = act(dot(X[row,:K], w) + b);  act: 0 none, 2 sigmoid
__global__ void k_rowdot(const float* __restrict__ X, int ldx,
                         const float* __restrict__ w, const float* __restrict__ bias,
                         float* __restrict__ out, int K, int act) {
  long r = blockIdx.x;
  float s = 0.f;
  for (int c = threadIdx.x; c < K; c += blockDim.x) s += X[r * ldx + c] * w[c];
  s = block_sum(s);
  if (threadIdx.x == 0) {
    s += bias[0];
    if (act == 2) s = 1.f / (1.f + expf(-s));
    out[r] = s;
  }
}

// CIF: normalize alphas to sum L, integrate-and-fire, emit first 128 indices.
__global__ void k_cif(const float* __restrict__ alpha, int T,
                      const int* __restrict__ tl, int* __restrict__ idx) {
  float s = 0.f;
  for (int t = threadIdx.x; t < T; t += blockDim.x) s += alpha[t];
  s = block_sum(s);
  if (threadIdx.x == 0) {
    int L = tl[0];
    if (L > 128) L = 128;
    if (L < 1) L = 1;
    float sc = (float)L / (s + 1e-8f);
    float integ = 0.f;
    int c = 0;
    for (int t = 0; t < T; ++t) {
      integ += alpha[t] * sc;
      if (integ >= 1.f) {
        integ -= 1.f;
        if (c < 128) idx[c++] = t;
      }
    }
    while (c < 128) idx[c++] = T - 1;
  }
}

// sampled = enc[idx[row]] + PE(row); f32 + f16
__global__ void k_gather_pe(const float* __restrict__ enc, const int* __restrict__ idx,
                            float* __restrict__ out, h16* __restrict__ outh, int D) {
  long r = blockIdx.x;
  long src = idx[r];
  for (int c = threadIdx.x; c < D; c += blockDim.x) {
    float v = enc[src * D + c] + pe_val((int)r, c, D);
    out[r * D + c] = v;
    outh[r * D + c] = (h16)v;
  }
}

// hx[row] = emb[ids[row]] + PE(row % 8)
__global__ void k_embed(const int* __restrict__ ids, const float* __restrict__ emb,
                        float* __restrict__ out, h16* __restrict__ outh, int D) {
  long r = blockIdx.x;
  long e = ids[r];
  int pos = (int)(r % 8);
  for (int c = threadIdx.x; c < D; c += blockDim.x) {
    float v = emb[e * D + c] + pe_val(pos, c, D);
    out[r * D + c] = v;
    outh[r * D + c] = (h16)v;
  }
}

// Tiny hotword self-attention: 50 seqs x 8 tokens, 4 heads of dk=128.
// grid=(50,4), block=128. Q,K,V,ctx: [400,512] f32.
__global__ void k_small_attn(const float* __restrict__ Q, const float* __restrict__ K,
                             const float* __restrict__ V, float* __restrict__ ctx,
                             int D, int nh) {
  int s = blockIdx.x, h = blockIdx.y;
  int dk = D / nh;  // 128
  __shared__ float q[8][128], k[8][128], v[8][128], a[8][8];
  int tid = threadIdx.x;
  for (int i = tid; i < 8 * dk; i += blockDim.x) {
    int t = i / dk, d = i % dk;
    long g = (long)(s * 8 + t) * D + h * dk + d;
    q[t][d] = Q[g]; k[t][d] = K[g]; v[t][d] = V[g];
  }
  __syncthreads();
  if (tid < 64) {
    int i = tid >> 3, j = tid & 7;
    float sum = 0.f;
    for (int d = 0; d < dk; ++d) sum += q[i][d] * k[j][d];
    a[i][j] = sum / sqrtf((float)dk);
  }
  __syncthreads();
  if (tid < 8) {
    float mx = -1e30f;
    for (int j = 0; j < 8; ++j) mx = fmaxf(mx, a[tid][j]);
    float ssum = 0.f;
    for (int j = 0; j < 8; ++j) { float e = expf(a[tid][j] - mx); a[tid][j] = e; ssum += e; }
    float inv = 1.f / ssum;
    for (int j = 0; j < 8; ++j) a[tid][j] *= inv;
  }
  __syncthreads();
  for (int i = tid; i < 8 * dk; i += blockDim.x) {
    int t = i / dk, d = i % dk;
    float sum = 0.f;
    for (int j = 0; j < 8; ++j) sum += a[t][j] * v[j][d];
    ctx[(long)(s * 8 + t) * D + h * dk + d] = sum;
  }
}

// masked mean-pool over 8 tokens per sequence: hx[400,D] -> hw[50,D] (+f16)
__global__ void k_maskpool(const float* __restrict__ hx, const float* __restrict__ mask,
                           float* __restrict__ hw, h16* __restrict__ hwh, int D) {
  long s = blockIdx.x;
  float den = 0.f;
  for (int t = 0; t < 8; ++t) den += mask[s * 8 + t];
  float inv = 1.f / (den + 1e-8f);
  for (int c = threadIdx.x; c < D; c += blockDim.x) {
    float num = 0.f;
    for (int t = 0; t < 8; ++t) num += hx[(s * 8 + t) * D + c] * mask[s * 8 + t];
    float v = num * inv;
    hw[s * D + c] = v;
    hwh[s * D + c] = (h16)v;
  }
}

// gate[row] = sigmoid(dot(w[0:D],y[row]) + dot(w[D:2D],ctx[row]) + b)
__global__ void k_gate(const float* __restrict__ y, const float* __restrict__ ctx,
                       const float* __restrict__ w, const float* __restrict__ b,
                       float* __restrict__ gate, int D) {
  long r = blockIdx.x;
  float s = 0.f;
  for (int c = threadIdx.x; c < D; c += blockDim.x)
    s += w[c] * y[r * D + c] + w[D + c] * ctx[r * D + c];
  s = block_sum(s);
  if (threadIdx.x == 0) gate[r] = 1.f / (1.f + expf(-(s + b[0])));
}

// x[row] *= g[row]; write f16 mirror
__global__ void k_scale_cvt(float* __restrict__ x, const float* __restrict__ g,
                            h16* __restrict__ xh, int D) {
  long r = blockIdx.x;
  float gv = g[r];
  for (int c = threadIdx.x; c < D; c += blockDim.x) {
    float v = x[r * D + c] * gv;
    x[r * D + c] = v;
    xh[r * D + c] = (h16)v;
  }
}

// out = logits + w[row] * bias_logits
__global__ void k_final(const float* __restrict__ logits, const float* __restrict__ bl,
                        const float* __restrict__ w, float* __restrict__ out, int V) {
  long r = blockIdx.x;
  float wv = w[r];
  for (int c = threadIdx.x; c < V; c += blockDim.x)
    out[r * V + c] = logits[r * V + c] + wv * bl[r * V + c];
}

// ---------------------------------------------------------------------------
// Host orchestration
// ---------------------------------------------------------------------------
struct Lyr {
  const float *wq, *bq, *wk, *bk, *wv, *bv, *wo, *bo;
  const float *w1, *b1, *w2, *b2, *g1, *e1, *g2, *e2;
};
struct LyrW { const h16 *wq, *wk, *wv, *wo, *w1, *w2; };

extern "C" void kernel_launch(void* const* d_in, const int* in_sizes, int n_in,
                              void* d_out, int out_size, void* d_ws, size_t ws_size,
                              hipStream_t stream) {
  (void)in_sizes; (void)n_in; (void)out_size; (void)ws_size;

  // ---- model constants ----
  const int T = 1500, Tp = 1504, D = 512, DFF = 2048, V = 8000, L = 128;

  // ---- input cursor (setup_inputs order, params flattened depth-first) ----
  const float* speech  = (const float*)d_in[0];
  const int*   hwids   = (const int*)d_in[1];
  const float* hwmask  = (const float*)d_in[2];
  const int*   tlen    = (const int*)d_in[3];
  int q = 4;
  auto nf = [&]() { return (const float*)d_in[q++]; };
  auto rdLyr = [&](Lyr& Lr) {
    Lr.wq = nf(); Lr.bq = nf(); Lr.wk = nf(); Lr.bk = nf();
    Lr.wv = nf(); Lr.bv = nf(); Lr.wo = nf(); Lr.bo = nf();
    Lr.w1 = nf(); Lr.b1 = nf(); Lr.w2 = nf(); Lr.b2 = nf();
    Lr.g1 = nf(); Lr.e1 = nf(); Lr.g2 = nf(); Lr.e2 = nf();
  };

  const float* enc_in_w = nf(); const float* enc_in_b = nf();
  Lyr encL[6]; for (int i = 0; i < 6; ++i) rdLyr(encL[i]);
  const float* c1w = nf(); const float* c1b = nf();
  const float* c2w = nf(); const float* c2b = nf();
  const float* fcw = nf(); const float* fcb = nf();
  Lyr decL[6]; for (int i = 0; i < 6; ++i) rdLyr(decL[i]);
  const float* dow = nf(); const float* dob = nf();
  const float* emb = nf();
  Lyr hwlL[2]; for (int i = 0; i < 2; ++i) rdLyr(hwlL[i]);
  const float* awq = nf(); const float* abq = nf();
  const float* awk = nf(); const float* abk = nf();
  const float* awv = nf(); const float* abv = nf();
  const float* awo = nf(); const float* abo = nf();
  const float* gw = nf();  const float* gb = nf();
  const float* bpw = nf(); const float* bpb = nf();
  const float* w1w = nf(); const float* w1b = nf();
  const float* w2w = nf(); const float* w2b = nf();

  // ---- bump allocator over workspace ----
  char* base = (char*)d_ws;
  size_t off = 0;
  auto alloc = [&](size_t bytes) -> void* {
    void* p = base + off;
    off += (bytes + 255) & ~(size_t)255;
    return p;
  };

  // ---- weight prep: transpose + f16 convert (Bt layout [N,Kp]) ----
  auto prepW = [&](const float* W, int K, int N, int Kp) -> const h16* {
    h16* Wt = (h16*)alloc((size_t)N * Kp * sizeof(h16));
    long n = (long)N * Kp;
    k_tcvt<<<dim3((unsigned)((n + 255) / 256)), 256, 0, stream>>>(W, Wt, K, N, Kp, n);
    return Wt;
  };
  auto prepL = [&](const Lyr& Lr) -> LyrW {
    LyrW w;
    w.wq = prepW(Lr.wq, D, D, D);   w.wk = prepW(Lr.wk, D, D, D);
    w.wv = prepW(Lr.wv, D, D, D);   w.wo = prepW(Lr.wo, D, D, D);
    w.w1 = prepW(Lr.w1, D, DFF, D); w.w2 = prepW(Lr.w2, DFF, D, DFF);
    return w;
  };
  const h16* encin_t = prepW(enc_in_w, 80, D, 96);
  LyrW encW[6]; for (int i = 0; i < 6; ++i) encW[i] = prepL(encL[i]);
  LyrW decW[6]; for (int i = 0; i < 6; ++i) decW[i] = prepL(decL[i]);
  LyrW hwW[2];  for (int i = 0; i < 2; ++i) hwW[i] = prepL(hwlL[i]);
  const h16* dow_t = prepW(dow, D, V, D);
  const h16* aq_t = prepW(awq, D, D, D);
  const h16* ak_t = prepW(awk, D, D, D);
  const h16* av_t = prepW(awv, D, D, D);
  const h16* ao_t = prepW(awo, D, D, D);
  const h16* bp_t = prepW(bpw, D, V, D);
  const h16* w1_t = prepW(w1w, D, 128, D);

  // ---- shared activation scratch (sized for the encoder, reused) ----
  float* Qf  = (float*)alloc((size_t)Tp * D * 4); h16* Qh = (h16*)alloc((size_t)Tp * D * 2);
  float* Kf  = (float*)alloc((size_t)Tp * D * 4); h16* Kh = (h16*)alloc((size_t)Tp * D * 2);
  float* Vf  = (float*)alloc((size_t)Tp * D * 4);
  h16*  Vt  = (h16*)alloc((size_t)D * Tp * 2);
  float* S   = (float*)alloc((size_t)8 * T * Tp * 4);
  h16*  P   = (h16*)alloc((size_t)8 * T * Tp * 2);
  float* Cx  = (float*)alloc((size_t)Tp * D * 4); h16* Cxh = (h16*)alloc((size_t)Tp * D * 2);
  float* Of  = (float*)alloc((size_t)Tp * D * 4);
  float* Xa  = (float*)alloc((size_t)Tp * D * 4); h16* Xah = (h16*)alloc((size_t)Tp * D * 2);
  float* Ff  = (float*)alloc((size_t)Tp * DFF * 4); h16* Fh = (h16*)alloc((size_t)Tp * DFF * 2);
  float* F2  = (float*)alloc((size_t)Tp * D * 4);

  auto gemm = [&](const h16* A, int lda, long sA, const h16* B, int ldb, long sB,
                  const float* bias, float* C, int ldc, long sC,
                  h16* Ch, int ldch, long sCh, int M, int N, int K, int act, int batch) {
    dim3 g((unsigned)((N + 63) / 64), (unsigned)((M + 127) / 128), (unsigned)batch);
    k_gemm<<<g, dim3(128), 0, stream>>>(A, lda, sA, B, ldb, sB, bias,
                                        C, ldc, sC, Ch, ldch, sCh, M, N, K, act);
  };

  // Full transformer layer with WMMA attention (heads batched via grid.z).
  auto runLayer = [&](float* x, h16* xh, const Lyr& Lr, const LyrW& W,
                      int R, int Rp, int nh) {
    int dk = D / nh;
    float iscale = 1.f / sqrtf((float)dk);
    gemm(xh, D, 0, W.wq, D, 0, Lr.bq, Qf, D, 0, Qh, D, 0, R, D, D, 0, 1);
    gemm(xh, D, 0, W.wk, D, 0, Lr.bk, Kf, D, 0, Kh, D, 0, R, D, D, 0, 1);
    gemm(xh, D, 0, W.wv, D, 0, Lr.bv, Vf, D, 0, nullptr, 0, 0, R, D, D, 0, 1);
    { long n = (long)D * Rp;
      k_tcvt<<<dim3((unsigned)((n + 255) / 256)), 256, 0, stream>>>(Vf, Vt, R, D, Rp, n); }
    // scores: per head  S[h] = Q_h * K_h^T   (M=R,N=R,K=dk)
    gemm(Qh, D, dk, Kh, D, dk, nullptr, S, Rp, (long)R * Rp,
         nullptr, 0, 0, R, R, dk, 0, nh);
    k_softmax<<<dim3((unsigned)R, (unsigned)nh), 256, 0, stream>>>(
        S, (long)R * Rp, Rp, P, (long)R * Rp, Rp, R, iscale);
    // ctx: per head  C[:, h*dk : h*dk+dk] = P[h] * V_h   (K=Rp zero-padded)
    gemm(P, Rp, (long)R * Rp, Vt, Rp, (long)dk * Rp, nullptr,
         Cx, D, dk, Cxh, D, dk, R, dk, Rp, 0, nh);
    gemm(Cxh, D, 0, W.wo, D, 0, Lr.bo, Of, D, 0, nullptr, 0, 0, R, D, D, 0, 1);
    k_ln_res<<<dim3((unsigned)R), 256, 0, stream>>>(x, Of, Lr.g1, Lr.e1, Xa, Xah, D);
    gemm(Xah, D, 0, W.w1, D, 0, Lr.b1, Ff, DFF, 0, Fh, DFF, 0, R, DFF, D, 1, 1);
    gemm(Fh, DFF, 0, W.w2, DFF, 0, Lr.b2, F2, D, 0, nullptr, 0, 0, R, D, DFF, 0, 1);
    k_ln_res<<<dim3((unsigned)R), 256, 0, stream>>>(Xa, F2, Lr.g2, Lr.e2, x, xh, D);
  };

  // Hotword layer: GEMM projections + tiny per-sequence attention.
  auto runHwLayer = [&](float* x, h16* xh, const Lyr& Lr, const LyrW& W) {
    const int R = 400;
    gemm(xh, D, 0, W.wq, D, 0, Lr.bq, Qf, D, 0, nullptr, 0, 0, R, D, D, 0, 1);
    gemm(xh, D, 0, W.wk, D, 0, Lr.bk, Kf, D, 0, nullptr, 0, 0, R, D, D, 0, 1);
    gemm(xh, D, 0, W.wv, D, 0, Lr.bv, Vf, D, 0, nullptr, 0, 0, R, D, D, 0, 1);
    k_small_attn<<<dim3(50, 4), 128, 0, stream>>>(Qf, Kf, Vf, Cx, D, 4);
    { long n = (long)R * D;
      k_cvt<<<dim3((unsigned)((n + 255) / 256)), 256, 0, stream>>>(Cx, Cxh, n); }
    gemm(Cxh, D, 0, W.wo, D, 0, Lr.bo, Of, D, 0, nullptr, 0, 0, R, D, D, 0, 1);
    k_ln_res<<<dim3((unsigned)R), 256, 0, stream>>>(x, Of, Lr.g1, Lr.e1, Xa, Xah, D);
    gemm(Xah, D, 0, W.w1, D, 0, Lr.b1, Ff, DFF, 0, Fh, DFF, 0, R, DFF, D, 1, 1);
    gemm(Fh, DFF, 0, W.w2, DFF, 0, Lr.b2, F2, D, 0, nullptr, 0, 0, R, D, DFF, 0, 1);
    k_ln_res<<<dim3((unsigned)R), 256, 0, stream>>>(Xa, F2, Lr.g2, Lr.e2, x, xh, D);
  };

  // ===== Encoder =====
  h16* sp16 = (h16*)alloc((size_t)T * 96 * 2);
  { long n = (long)T * 96;
    k_cvt_pad<<<dim3((unsigned)((n + 255) / 256)), 256, 0, stream>>>(speech, sp16, 80, 96, n); }
  float* X = (float*)alloc((size_t)T * D * 4);
  h16*  Xh = (h16*)alloc((size_t)T * D * 2);
  gemm(sp16, 96, 0, encin_t, 96, 0, enc_in_b, X, D, 0, nullptr, 0, 0, T, D, 96, 0, 1);
  k_add_pe_cvt<<<dim3((unsigned)T), 256, 0, stream>>>(X, Xh, D, T);
  for (int i = 0; i < 6; ++i) runLayer(X, Xh, encL[i], encW[i], T, Tp, 8);

  // ===== Predictor + CIF =====
  float* H1 = (float*)alloc((size_t)T * 256 * 4);
  float* H2 = (float*)alloc((size_t)T * 256 * 4);
  k_conv1d_relu<<<dim3((unsigned)((T * 256 + 255) / 256)), 256, 0, stream>>>(
      X, c1w, c1b, H1, T, 512, 256);
  k_conv1d_relu<<<dim3((unsigned)((T * 256 + 255) / 256)), 256, 0, stream>>>(
      H1, c2w, c2b, H2, T, 256, 256);
  float* alph = (float*)alloc((size_t)T * 4);
  k_rowdot<<<dim3((unsigned)T), 128, 0, stream>>>(H2, 256, fcw, fcb, alph, 256, 2);
  int* idxb = (int*)alloc(128 * sizeof(int));
  k_cif<<<1, 256, 0, stream>>>(alph, T, tlen, idxb);

  // ===== Decoder =====
  float* Y = (float*)alloc((size_t)L * D * 4);
  h16*  Yh = (h16*)alloc((size_t)L * D * 2);
  k_gather_pe<<<dim3((unsigned)L), 256, 0, stream>>>(X, idxb, Y, Yh, D);
  for (int i = 0; i < 6; ++i) runLayer(Y, Yh, decL[i], decW[i], L, L, 8);
  float* logits = (float*)alloc((size_t)L * V * 4);
  gemm(Yh, D, 0, dow_t, D, 0, dob, logits, V, 0, nullptr, 0, 0, L, V, D, 0, 1);

  // ===== Hotword branch =====
  float* HX = (float*)alloc((size_t)400 * D * 4);
  h16*  HXh = (h16*)alloc((size_t)400 * D * 2);
  k_embed<<<dim3(400), 256, 0, stream>>>(hwids, emb, HX, HXh, D);
  for (int i = 0; i < 2; ++i) runHwLayer(HX, HXh, hwlL[i], hwW[i]);
  float* HW = (float*)alloc((size_t)50 * D * 4);
  h16*  HWh = (h16*)alloc((size_t)50 * D * 2);
  k_maskpool<<<dim3(50), 256, 0, stream>>>(HX, hwmask, HW, HWh, D);

  // ===== Cross attention (4 heads, dk=128, kv len 50 padded to 64) =====
  const int HHn = 4, hdk = 128, KV = 50, KVp = 64;
  gemm(Yh, D, 0, aq_t, D, 0, abq, Qf, D, 0, Qh, D, 0, L, D, D, 0, 1);
  gemm(HWh, D, 0, ak_t, D, 0, abk, Kf, D, 0, Kh, D, 0, KV, D, D, 0, 1);
  gemm(HWh, D, 0, av_t, D, 0, abv, Vf, D, 0, nullptr, 0, 0, KV, D, D, 0, 1);
  { long n = (long)D * KVp;
    k_tcvt<<<dim3((unsigned)((n + 255) / 256)), 256, 0, stream>>>(Vf, Vt, KV, D, KVp, n); }
  gemm(Qh, D, hdk, Kh, D, hdk, nullptr, S, KVp, (long)L * KVp,
       nullptr, 0, 0, L, KV, hdk, 0, HHn);
  k_softmax<<<dim3((unsigned)L, (unsigned)HHn), 256, 0, stream>>>(
      S, (long)L * KVp, KVp, P, (long)L * KVp, KVp, KV, 1.f / sqrtf((float)hdk));
  gemm(P, KVp, (long)L * KVp, Vt, KVp, (long)hdk * KVp, nullptr,
       Cx, D, hdk, Cxh, D, hdk, L, hdk, KVp, 0, HHn);
  float* CtxO = (float*)alloc((size_t)L * D * 4);
  h16*  CtxOh = (h16*)alloc((size_t)L * D * 2);
  gemm(Cxh, D, 0, ao_t, D, 0, abo, CtxO, D, 0, nullptr, 0, 0, L, D, D, 0, 1);

  // gate, bias logits, wnet, final blend
  float* gate = (float*)alloc((size_t)L * 4);
  k_gate<<<dim3((unsigned)L), 256, 0, stream>>>(Y, CtxO, gw, gb, gate, D);
  k_scale_cvt<<<dim3((unsigned)L), 256, 0, stream>>>(CtxO, gate, CtxOh, D);
  float* blog = (float*)alloc((size_t)L * V * 4);
  gemm(CtxOh, D, 0, bp_t, D, 0, bpb, blog, V, 0, nullptr, 0, 0, L, V, D, 0, 1);
  float* W1 = (float*)alloc((size_t)L * 128 * 4);
  gemm(CtxOh, D, 0, w1_t, D, 0, w1b, W1, 128, 0, nullptr, 0, 0, L, 128, D, 1, 1);
  float* wvec = (float*)alloc((size_t)L * 4);
  k_rowdot<<<dim3((unsigned)L), 128, 0, stream>>>(W1, 128, w2w, w2b, wvec, 128, 2);
  k_final<<<dim3((unsigned)L), 256, 0, stream>>>(logits, blog, wvec, (float*)d_out, V);
}
